// BlockPAC_24842090840210
// MI455X (gfx1250) — compile-verified, matrix operand order checked by reading, not run on hardware
//
#include <hip/hip_runtime.h>

typedef _Float16 f16;
typedef __attribute__((ext_vector_type(16))) _Float16 v16h;
typedef __attribute__((ext_vector_type(8)))  _Float16 v8h;
typedef __attribute__((ext_vector_type(8)))  float    v8f;

#define NB    8
#define CIN   64
#define COUT  64
#define CG    16
#define HH_   128
#define WW_   128
#define KTOT  576      /* CIN * 9 */
#define KSTEPS 18      /* KTOT / 32 */
#define LDSS  584      /* 576 + 8 halves pad: stride 292 dwords, gcd(36,64)=4 -> conflict-free rows */

// ---- one-time weight transpose + f16 convert: Bt[o][tap*64+c] = (f16)w[o][c][tap] ----
__global__ __launch_bounds__(256) void pac_convert_weights(const float* __restrict__ w,
                                                           f16* __restrict__ Bt) {
    int idx = blockIdx.x * 256 + threadIdx.x;
    if (idx >= COUT * KTOT) return;
    int o  = idx / KTOT;
    int r  = idx - o * KTOT;
    int p  = r >> 6;     // tap 0..8
    int c  = r & 63;     // in-channel
    Bt[idx] = (f16)w[(o * CIN + c) * 9 + p];
}

// ---- main PAC kernel: guide-kernel -> modulated im2col (LDS f16) -> WMMA GEMM ----
__global__ __launch_bounds__(256) void pac_wmma_kernel(const float* __restrict__ x,
                                                       const float* __restrict__ guide,
                                                       const f16*   __restrict__ Bt,
                                                       const float* __restrict__ bias,
                                                       float*       __restrict__ out) {
    __shared__ f16   Alds[32 * LDSS];     // 32 pixels x 576 K, f16
    __shared__ float kernw[32][12];       // per-pixel per-tap Gaussian weight

    const int t      = threadIdx.x;
    const int bx     = blockIdx.x;        // 4096 blocks = NB * HH_ * (WW_/32)
    const int tile_w = bx & 3;
    const int h      = (bx >> 2) & 127;
    const int b      = bx >> 9;
    const int w0     = tile_w * 32;

    const float* gb = guide + (size_t)b * CG  * HH_ * WW_;
    const float* xb = x     + (size_t)b * CIN * HH_ * WW_;

    // ---------- phase 1: Gaussian kernel weights from guide ----------
    for (int task = t; task < 32 * 9; task += 256) {
        int pixel = task / 9;
        int tap   = task - pixel * 9;
        int di = tap / 3 - 1, dj = tap % 3 - 1;
        int w  = w0 + pixel;
        int hh = h + di, ww = w + dj;
        bool inb = (hh >= 0) & (hh < HH_) & (ww >= 0) & (ww < WW_);
        float s = 0.f;
        #pragma unroll
        for (int c = 0; c < CG; ++c) {
            float gc = gb[(size_t)c * HH_ * WW_ + h * WW_ + w];
            float gt = inb ? gb[(size_t)c * HH_ * WW_ + hh * WW_ + ww] : 0.f; // zero-pad semantics
            float d  = gt - gc;
            s += d * d;
        }
        kernw[pixel][tap] = __expf(-0.5f * s);
    }
    __syncthreads();

    // ---------- phase 2: build modulated im2col A-tile in LDS (f16) ----------
    {
        const int c = t >> 2;   // 0..63 input channel (coalesced row reads per channel)
        const int q = t & 3;    // pixel octet 0..3
        #pragma unroll
        for (int tap = 0; tap < 9; ++tap) {
            const int di = tap / 3 - 1, dj = tap % 3 - 1;
            const int hr = h + di;
            const bool hin = (hr >= 0) & (hr < HH_);
            const float* xrow = xb + (size_t)c * HH_ * WW_ + (size_t)hr * WW_;
            const int kcol = tap * 64 + c;
            #pragma unroll
            for (int i = 0; i < 8; ++i) {
                int pixel = q * 8 + i;
                int ww = w0 + pixel + dj;
                float v = (hin && ww >= 0 && ww < WW_) ? xrow[ww] : 0.f;
                v *= kernw[pixel][tap];
                Alds[pixel * LDSS + kcol] = (f16)v;
            }
        }
    }
    __syncthreads();

    // ---------- phase 3: WMMA GEMM  D[16 pix x 16 cout] per wave ----------
    const int wave   = t >> 5;            // wave32
    const int lane   = t & 31;
    const int m_base = (wave >> 2) * 16;  // pixel sub-tile
    const int n0     = (wave & 3) * 16;   // cout sub-tile
    const int rown   = lane & 15;
    const int khalf  = (lane >> 4) * 16;  // K-half: lane half h holds k = 16h..16h+15 (contiguous order, same for A and B)
    const f16* bptr  = Bt + (size_t)(n0 + rown) * KTOT + khalf;
    const f16* aptr  = &Alds[(m_base + rown) * LDSS + khalf];

    v8f acc = {};
    #pragma unroll
    for (int kk = 0; kk < KSTEPS; ++kk) {
        v8h alo = *(const v8h*)(aptr + kk * 32);
        v8h ahi = *(const v8h*)(aptr + kk * 32 + 8);
        v16h a  = __builtin_shufflevector(alo, ahi, 0,1,2,3,4,5,6,7,8,9,10,11,12,13,14,15);
        v8h blo = *(const v8h*)(bptr + kk * 32);
        v8h bhi = *(const v8h*)(bptr + kk * 32 + 8);
        v16h bm = __builtin_shufflevector(blo, bhi, 0,1,2,3,4,5,6,7,8,9,10,11,12,13,14,15);
        acc = __builtin_amdgcn_wmma_f32_16x16x32_f16(false, a, false, bm,
                                                     (short)0, acc, false, false);
    }

    // ---------- bias + ReLU + store (two aligned float4 per lane) ----------
    const int o  = n0 + rown;
    const float bv = bias[o];
    const int p0 = m_base + 8 * (lane >> 4);           // C layout: M = r + 8*(lane>>4)
    float* op = out + (((size_t)b * COUT + o) * HH_ + h) * WW_ + w0 + p0;
    float4 s0, s1;
    s0.x = fmaxf(acc[0] + bv, 0.f);
    s0.y = fmaxf(acc[1] + bv, 0.f);
    s0.z = fmaxf(acc[2] + bv, 0.f);
    s0.w = fmaxf(acc[3] + bv, 0.f);
    s1.x = fmaxf(acc[4] + bv, 0.f);
    s1.y = fmaxf(acc[5] + bv, 0.f);
    s1.z = fmaxf(acc[6] + bv, 0.f);
    s1.w = fmaxf(acc[7] + bv, 0.f);
    *(float4*)op       = s0;
    *(float4*)(op + 4) = s1;
}

extern "C" void kernel_launch(void* const* d_in, const int* in_sizes, int n_in,
                              void* d_out, int out_size, void* d_ws, size_t ws_size,
                              hipStream_t stream) {
    const float* x      = (const float*)d_in[0];
    const float* guide  = (const float*)d_in[1];
    const float* weight = (const float*)d_in[2];
    const float* bias   = (const float*)d_in[3];
    float* out = (float*)d_out;
    f16* Bt = (f16*)d_ws;                  // needs COUT*KTOT*2 = 73,728 bytes

    pac_convert_weights<<<(COUT * KTOT + 255) / 256, 256, 0, stream>>>(weight, Bt);
    // 8 batches * 128 rows * 4 width-tiles = 4096 workgroups, 256 threads (8 wave32)
    pac_wmma_kernel<<<NB * HH_ * (WW_ / 32), 256, 0, stream>>>(x, guide, Bt, bias, out);
}
